// SSMLayer_74577812128336
// MI455X (gfx1250) — matrix operand order repaired and checked by reading, not compiled, not run on hardware
//
#include <hip/hip_runtime.h>

// ---------------------------------------------------------------------------
// SSM layer for MI455X (gfx1250, wave32).
//  - GEMMs via v_wmma_f32_16x16x32_bf16 (8-wave blocks, 16x16 tiles x 4 N-tiles)
//  - B fragments delivered by the Tensor Data Mover (tensor_load_to_lds,
//    TENSORcnt) into a ping-pong LDS buffer: DMA prefetches K-step kt+1 while
//    waves run WMMAs on kt. No VGPR staging traffic, no loadcnt stall.
//  - Length-4096 selective scan parallelized as 128-step chunks + carry scan.
// ---------------------------------------------------------------------------

typedef __attribute__((ext_vector_type(16))) __bf16 v16bf;
typedef __attribute__((ext_vector_type(8)))  float  v8f;
typedef __attribute__((ext_vector_type(4)))  unsigned int v4u;
typedef __attribute__((ext_vector_type(8)))  int    v8i_t;
typedef __attribute__((ext_vector_type(4)))  int    v4i_t;

#define TOKENS   32768      // BATCH * SEQ
#define DM       512
#define DN       64
#define SEQ_LEN  4096
#define NBATCH   8
#define CHUNK    128
#define NCHUNK   (SEQ_LEN / CHUNK)   // 32

union FragBF16 { uint4 u[2]; v16bf v; };

// ---------------------------------------------------------------------------
// TDM: 1-D tile load, global -> LDS. nbytes must be a multiple of 8.
// Descriptor per CDNA5 ISA:
//   group0: count=1 | lds_addr | global_addr[56:0] | type=2
//   group1: data_size=3 (8B), tensor_dim0=tile_dim0=n8, dim1=1, stride0=n8
// 6-arg builtin (amdgpu-toolchain/clang-23 signature):
//   (uint32x4 g0, int32x8 g1, int32x4 g2, int32x4 g3, int32x8 g4, i32 cpol)
// Issued by one wave; completion via s_wait_tensorcnt.
// ---------------------------------------------------------------------------
__device__ __forceinline__ void tdm_load_1d(const void* gsrc, unsigned lds_off,
                                            int nbytes) {
  unsigned long long ga = (unsigned long long)gsrc;
  int n8 = nbytes >> 3;
  v4u g0;
  g0[0] = 1u;                                            // count=1
  g0[1] = lds_off;                                       // LDS byte address
  g0[2] = (unsigned)(ga & 0xffffffffu);                  // addr[31:0]
  g0[3] = (unsigned)((ga >> 32) & 0x1ffffffu) | (2u << 30); // addr[56:32]|type=2
  v8i_t g1;
  g1[0] = (3 << 16);                                     // data_size=3 (8B)
  g1[1] = (n8 & 0xffff) << 16;                           // tensor_dim0[15:0]
  g1[2] = ((n8 >> 16) & 0xffff) | (1 << 16);             // dim0[31:16]|dim1 lo=1
  g1[3] = (n8 & 0xffff) << 16;                           // dim1 hi=0 | tile_dim0
  g1[4] = 1;                                             // tile_dim1=1
  g1[5] = n8;                                            // dim0_stride[31:0]
  g1[6] = 0;
  g1[7] = 0;
  v4i_t z4 = {0, 0, 0, 0};
  v8i_t z8 = {0, 0, 0, 0, 0, 0, 0, 0};
  __builtin_amdgcn_tensor_load_to_lds(g0, g1, z4, z4, z8, 0);
}

// ---------------------------------------------------------------------------
// Weight pack: row-major f32 W[K][Nsrc] -> bf16 WMMA B-fragment layout.
// B (32x16 bf16), contiguous k order used here: lane l holds column n = l&15, K = (l>>4)*16 + j.
// ---------------------------------------------------------------------------
__global__ __launch_bounds__(256) void pack_b_kernel(
    const float* __restrict__ W, __bf16* __restrict__ dst,
    int K, int Nsrc, int Ntot, int ncolOff)
{
  int id = blockIdx.x * 256 + threadIdx.x;
  if (id >= K * Nsrc) return;
  int k  = id / Nsrc;
  int nc = id % Nsrc;
  int n  = nc + ncolOff;
  int kt = k >> 5, kl = k & 31;
  int nt = n >> 4, nl = n & 15;
  int lane = ((kl >> 4) << 4) + nl;     // (kl/16)*16 + column
  int j    = kl & 15;
  size_t o = (((size_t)kt * (Ntot >> 4) + nt) * 32 + lane) * 16 + j;
  dst[o] = (__bf16)W[(size_t)k * Nsrc + nc];
}

// ---------------------------------------------------------------------------
// LayerNorm: one wave per token (512 elems, 16 per lane), write bf16 xn.
// ---------------------------------------------------------------------------
__global__ __launch_bounds__(256) void ln_kernel(
    const float* __restrict__ x, const float* __restrict__ gamma,
    const float* __restrict__ beta, __bf16* __restrict__ xn)
{
  int token = blockIdx.x * 8 + (threadIdx.x >> 5);
  int lane  = threadIdx.x & 31;
  const float4* row = (const float4*)(x + (size_t)token * DM);
  float4 v[4];
  float s = 0.f, sq = 0.f;
#pragma unroll
  for (int j = 0; j < 4; ++j) {
    v[j] = row[lane + 32 * j];
    s  += v[j].x + v[j].y + v[j].z + v[j].w;
    sq += v[j].x * v[j].x + v[j].y * v[j].y + v[j].z * v[j].z + v[j].w * v[j].w;
  }
#pragma unroll
  for (int m = 16; m > 0; m >>= 1) {
    s  += __shfl_xor(s,  m, 32);
    sq += __shfl_xor(sq, m, 32);
  }
  float mean = s * (1.0f / DM);
  float var  = sq * (1.0f / DM) - mean * mean;
  float rstd = rsqrtf(var + 1e-3f);
  __bf16* orow = xn + (size_t)token * DM;
#pragma unroll
  for (int j = 0; j < 4; ++j) {
    int c0 = (lane + 32 * j) * 4;
    union { __bf16 h[4]; uint2 u; } pk;
    pk.h[0] = (__bf16)((v[j].x - mean) * rstd * gamma[c0 + 0] + beta[c0 + 0]);
    pk.h[1] = (__bf16)((v[j].y - mean) * rstd * gamma[c0 + 1] + beta[c0 + 1]);
    pk.h[2] = (__bf16)((v[j].z - mean) * rstd * gamma[c0 + 2] + beta[c0 + 2]);
    pk.h[3] = (__bf16)((v[j].w - mean) * rstd * gamma[c0 + 3] + beta[c0 + 3]);
    *(uint2*)(orow + c0) = pk.u;
  }
}

// ---------------------------------------------------------------------------
// WMMA GEMM: C[M,N] = A[M,K](bf16,row-major) * Bpacked[K,N](bf16 fragments).
// Block = 256 threads = 8 waves; wave w -> M-tile (blockIdx.y*8+w), each wave
// covers 4 N-tiles (64 cols). Per K-step the 4 B fragments are one contiguous
// 4 KB region of the packed weights; the TDM streams them into the ping-pong
// LDS buffer (prefetch kt+1 during compute on kt). Wave 0 issues/drains the
// TDM op (TENSORcnt); the workgroup barrier publishes the tile.
// MODE epilogues:
//   0: +bias                  -> bf16 (u = xn@W_in + b_in)
//   1: raw f32                -> proj (u@[W_xs|W_B|W_C])
//   2: +extraF[n]*extraB[m,n] -> bf16 (ys@W_s2o + D*u)
//   3: +bias+extraF[m,n]      -> f32  (y1@W_out + b_out + residual)
// ---------------------------------------------------------------------------
template<int MODE>
__global__ __launch_bounds__(256) void gemm_kernel(
    const __bf16* __restrict__ A, const __bf16* __restrict__ Bp,
    int N, int K, int lda,
    const float*  __restrict__ bias,
    const float*  __restrict__ extraF,
    const __bf16* __restrict__ extraB,
    float* __restrict__ outF, __bf16* __restrict__ outB)
{
  constexpr int WN_ = 4;
  constexpr int TILE_BYTES = WN_ * 32 * 16 * 2;      // 4 KB per K-step
  __shared__ __bf16 ldsB[2][WN_ * 32 * 16];          // ping-pong

  const int wave  = threadIdx.x >> 5;
  const int lane  = threadIdx.x & 31;
  const int mtile = blockIdx.y * 8 + wave;
  const int mbase = mtile * 16;
  const int nt0   = blockIdx.x * WN_;
  const int NT    = N >> 4;
  const int KT    = K >> 5;

  v8f acc[WN_] = {};

  // contiguous packed-B source for this block's 4 N-tiles at K-step kt
  const size_t bstride = (size_t)NT * 32 * 16;       // elems per K-step
  const __bf16* bsrc0  = Bp + (size_t)nt0 * 32 * 16;
  const unsigned lds0  = (unsigned)(unsigned long long)(&ldsB[0][0]);
  const unsigned lds1  = (unsigned)(unsigned long long)(&ldsB[1][0]);

  // A-fragment addressing (16x32 bf16): lane holds row (lane&15),
  // K = {kh..kh+15}, kh = (lane>>4)*16  -> two 16B loads
  const int arow = mbase + (lane & 15);
  const int kh   = (lane >> 4) * 16;
  const __bf16* abase = A + (size_t)arow * lda + kh;

  // prologue: DMA K-step 0 into buffer 0
  if (wave == 0) {
    tdm_load_1d(bsrc0, lds0, TILE_BYTES);
    __builtin_amdgcn_s_wait_tensorcnt(0);
  }
  __syncthreads();

  for (int kt = 0; kt < KT; ++kt) {
    const int cur = kt & 1;
    // TDM-prefetch next K-step's B fragments into the other buffer
    if (wave == 0 && kt + 1 < KT) {
      tdm_load_1d(bsrc0 + (size_t)(kt + 1) * bstride,
                  cur ? lds0 : lds1, TILE_BYTES);
    }

    FragBF16 a;
    const __bf16* ap = abase + kt * 32;
    a.u[0] = *(const uint4*)ap;
    a.u[1] = *(const uint4*)(ap + 8);

    // load all 4 B fragments first (clause the LDS loads), then issue WMMAs
    FragBF16 b[WN_];
#pragma unroll
    for (int f = 0; f < WN_; ++f) {
      const __bf16* lp = ldsB[cur] + (f * 32 + lane) * 16;
      b[f].u[0] = *(const uint4*)lp;
      b[f].u[1] = *(const uint4*)(lp + 8);
    }
#pragma unroll
    for (int f = 0; f < WN_; ++f) {
      acc[f] = __builtin_amdgcn_wmma_f32_16x16x32_bf16(
          false, a.v, false, b[f].v, (short)0, acc[f], false, false);
    }

    if (wave == 0) __builtin_amdgcn_s_wait_tensorcnt(0);  // drain prefetch
    __syncthreads();   // publish ldsB[cur^1]; reads of ldsB[cur] complete
  }

  // C/D layout: lane l, vgpr r -> m = r + 8*(l>>4), n = l&15
  const int mrow = mbase + ((lane >> 4) << 3);
  const int ncol = lane & 15;
#pragma unroll
  for (int f = 0; f < WN_; ++f) {
    const int n = (nt0 + f) * 16 + ncol;
#pragma unroll
    for (int r = 0; r < 8; ++r) {
      const size_t o = (size_t)(mrow + r) * N + n;
      float vv = acc[f][r];
      if constexpr (MODE == 0) {
        vv += bias[n];
        outB[o] = (__bf16)vv;
      } else if constexpr (MODE == 1) {
        outF[o] = vv;
      } else if constexpr (MODE == 2) {
        vv += extraF[n] * (float)extraB[(size_t)(mrow + r) * DM + n];
        outB[o] = (__bf16)vv;
      } else {
        vv += bias[n] + extraF[o];
        outF[o] = vv;
      }
    }
  }
}

// ---------------------------------------------------------------------------
// Scan pass 1: per (batch, chunk, state) thread, local scan of 128 steps.
// Fuses Bu = xs * (Bm + b_B) from the proj buffer.
// ---------------------------------------------------------------------------
__global__ __launch_bounds__(256) void scan_chunks_kernel(
    const float* __restrict__ proj, const float* __restrict__ A_log,
    const float* __restrict__ b_B,
    float* __restrict__ hloc, float* __restrict__ carry)
{
  int t = blockIdx.x * 256 + threadIdx.x;      // 8*32*64 = 16384 threads
  int n = t & 63;
  int c = (t >> 6) & (NCHUNK - 1);
  int b = t >> 11;
  float a  = __expf(-__expf(A_log[n]));
  float bb = b_B[n];
  float h  = 0.f;
  int base = b * SEQ_LEN + c * CHUNK;
#pragma unroll 4
  for (int i = 0; i < CHUNK; ++i) {
    const float* pr = proj + (size_t)(base + i) * 192;
    __builtin_prefetch(pr + 192 * 2, 0, 1);
    float bu = pr[n] * (pr[64 + n] + bb);
    h = fmaf(a, h, bu);
    hloc[(size_t)(base + i) * DN + n] = h;
  }
  carry[(size_t)(b * NCHUNK + c) * DN + n] = h;
}

// ---------------------------------------------------------------------------
// Scan pass 2: serial scan of 32 chunk carries per (batch, state). In-place:
// carry[c] becomes the *incoming* global state for chunk c.
// ---------------------------------------------------------------------------
__global__ __launch_bounds__(256) void scan_carry_kernel(
    float* __restrict__ carry, const float* __restrict__ A_log)
{
  int t = blockIdx.x * 256 + threadIdx.x;      // 512 threads
  if (t >= NBATCH * DN) return;
  int n = t & 63;
  int b = t >> 6;
  float a = __expf(-__expf(A_log[n]));
  float aL = a;
#pragma unroll
  for (int i = 0; i < 7; ++i) aL *= aL;        // a^128
  float cin = 0.f;
  for (int c = 0; c < NCHUNK; ++c) {
    size_t idx = (size_t)(b * NCHUNK + c) * DN + n;
    float e = carry[idx];
    carry[idx] = cin;
    cin = fmaf(aL, cin, e);
  }
}

// ---------------------------------------------------------------------------
// Scan pass 3: fixup h = hloc + carry_in * a^(i+1); ys = (C + b_C) * h -> bf16
// ---------------------------------------------------------------------------
__global__ __launch_bounds__(256) void scan_fix_kernel(
    const float* __restrict__ hloc, const float* __restrict__ proj,
    const float* __restrict__ carry, const float* __restrict__ A_log,
    const float* __restrict__ b_C, __bf16* __restrict__ ys)
{
  size_t idx = (size_t)blockIdx.x * 256 + threadIdx.x;  // TOKENS*DN elems
  int n  = idx & 63;
  int sg = (int)(idx >> 6);          // global token
  int b  = sg >> 12;
  int s  = sg & (SEQ_LEN - 1);
  int c  = s >> 7;
  int i  = s & (CHUNK - 1);
  float a   = __expf(-__expf(A_log[n]));
  float cin = carry[(size_t)(b * NCHUNK + c) * DN + n];
  float h   = hloc[idx] + cin * __powf(a, (float)(i + 1));
  float Cv  = proj[(size_t)sg * 192 + 128 + n] + b_C[n];
  ys[idx] = (__bf16)(Cv * h);
}

// ---------------------------------------------------------------------------
// Launch
// ---------------------------------------------------------------------------
extern "C" void kernel_launch(void* const* d_in, const int* in_sizes, int n_in,
                              void* d_out, int out_size, void* d_ws, size_t ws_size,
                              hipStream_t stream) {
  (void)in_sizes; (void)n_in; (void)out_size; (void)ws_size;
  const float* x        = (const float*)d_in[0];
  const float* ln_gamma = (const float*)d_in[1];
  const float* ln_beta  = (const float*)d_in[2];
  const float* W_in     = (const float*)d_in[3];
  const float* b_in     = (const float*)d_in[4];
  const float* W_xs     = (const float*)d_in[5];
  const float* W_B      = (const float*)d_in[6];
  const float* b_B      = (const float*)d_in[7];
  const float* W_C      = (const float*)d_in[8];
  const float* b_C      = (const float*)d_in[9];
  const float* A_log    = (const float*)d_in[10];
  const float* Dvec     = (const float*)d_in[11];
  const float* W_s2o    = (const float*)d_in[12];
  const float* W_out    = (const float*)d_in[13];
  const float* b_out    = (const float*)d_in[14];
  float* out = (float*)d_out;

  size_t off = 0;
  auto carve = [&](size_t bytes) -> void* {
    void* p = (char*)d_ws + off;
    off += (bytes + 255) & ~(size_t)255;
    return p;
  };
  __bf16* wp_in  = (__bf16*)carve((size_t)DM * DM * 2);
  __bf16* wp_cat = (__bf16*)carve((size_t)DM * 192 * 2);
  __bf16* wp_s2o = (__bf16*)carve((size_t)DN * DM * 2);
  __bf16* wp_out = (__bf16*)carve((size_t)DM * DM * 2);
  __bf16* xn     = (__bf16*)carve((size_t)TOKENS * DM * 2);   // reused as y1
  __bf16* u      = (__bf16*)carve((size_t)TOKENS * DM * 2);
  float*  proj   = (float* )carve((size_t)TOKENS * 192 * 4);
  float*  hloc   = (float* )carve((size_t)TOKENS * DN * 4);
  float*  carry  = (float* )carve((size_t)NBATCH * NCHUNK * DN * 4);
  __bf16* ys     = (__bf16*)carve((size_t)TOKENS * DN * 2);
  __bf16* y1     = xn;   // xn is dead after the first GEMM

  // 1) pack weights into WMMA B-fragment layout (bf16)
  pack_b_kernel<<<(DM * DM + 255) / 256, 256, 0, stream>>>(W_in,  wp_in,  DM, DM, DM, 0);
  pack_b_kernel<<<(DM * DN + 255) / 256, 256, 0, stream>>>(W_xs,  wp_cat, DM, DN, 192, 0);
  pack_b_kernel<<<(DM * DN + 255) / 256, 256, 0, stream>>>(W_B,   wp_cat, DM, DN, 192, 64);
  pack_b_kernel<<<(DM * DN + 255) / 256, 256, 0, stream>>>(W_C,   wp_cat, DM, DN, 192, 128);
  pack_b_kernel<<<(DN * DM + 255) / 256, 256, 0, stream>>>(W_s2o, wp_s2o, DN, DM, DM, 0);
  pack_b_kernel<<<(DM * DM + 255) / 256, 256, 0, stream>>>(W_out, wp_out, DM, DM, DM, 0);

  // 2) LayerNorm -> bf16
  ln_kernel<<<TOKENS / 8, 256, 0, stream>>>(x, ln_gamma, ln_beta, xn);

  // 3) u = xn @ W_in + b_in            (M=32768, N=512, K=512)
  gemm_kernel<0><<<dim3(DM / 64, TOKENS / 128), 256, 0, stream>>>(
      xn, wp_in, DM, DM, DM, b_in, nullptr, nullptr, nullptr, u);

  // 4) proj = u @ [W_xs|W_B|W_C]       (N=192)
  gemm_kernel<1><<<dim3(192 / 64, TOKENS / 128), 256, 0, stream>>>(
      u, wp_cat, 192, DM, DM, nullptr, nullptr, nullptr, proj, nullptr);

  // 5) chunked selective scan
  scan_chunks_kernel<<<(NBATCH * NCHUNK * DN) / 256, 256, 0, stream>>>(
      proj, A_log, b_B, hloc, carry);
  scan_carry_kernel<<<2, 256, 0, stream>>>(carry, A_log);
  scan_fix_kernel<<<((size_t)TOKENS * DN) / 256, 256, 0, stream>>>(
      hloc, proj, carry, A_log, b_C, ys);

  // 6) y1 = ys @ W_s2o + D * u         (K=64)
  gemm_kernel<2><<<dim3(DM / 64, TOKENS / 128), 256, 0, stream>>>(
      ys, wp_s2o, DM, DN, DN, nullptr, Dvec, u, nullptr, y1);

  // 7) out = y1 @ W_out + b_out + x
  gemm_kernel<3><<<dim3(DM / 64, TOKENS / 128), 256, 0, stream>>>(
      y1, wp_out, DM, DM, DM, b_out, x, nullptr, out, nullptr);
}
